// MultiHeadAttention_23424751632866
// MI455X (gfx1250) — matrix operand order repaired and checked by reading, not compiled, not run on hardware
//
#include <hip/hip_runtime.h>
#include <hip/hip_bf16.h>

// MI455X / gfx1250 multi-head attention, bf16 WMMA + async-to-LDS pipeline.
// B=4, N=2048, E=1024, H=16, CH=64.

typedef __attribute__((ext_vector_type(16))) __bf16 v16bf;
typedef __attribute__((ext_vector_type(8)))  __bf16 v8bf;
typedef __attribute__((ext_vector_type(4)))  __bf16 v4bf;
typedef __attribute__((ext_vector_type(8)))  float  v8f;

#define DI __device__ __forceinline__

constexpr int B_  = 4;
constexpr int N_  = 2048;
constexpr int E_  = 1024;
constexpr int H_  = 16;
constexpr int CH_ = 64;

// ---------------- helpers ----------------

DI unsigned short f2bfu(float f) {           // fp32 -> bf16 bits, round-nearest-even
  unsigned int u = __float_as_uint(f);
  u += 0x7fffu + ((u >> 16) & 1u);
  return (unsigned short)(u >> 16);
}
DI __bf16 f2bf(float f) {
  unsigned short s = f2bfu(f);
  return __builtin_bit_cast(__bf16, s);
}

// A-operand (16x32 bf16, MxK): lane<16 holds K = kb+0..15 ; lane>=16 holds K = kb+16..31.
DI v16bf load_frag_A(const __bf16* row, int kb, int lhalf) {
  v8bf lo = *(const v8bf*)(row + kb + 16 * lhalf);
  v8bf hi = *(const v8bf*)(row + kb + 16 * lhalf + 8);
  return __builtin_shufflevector(lo, hi, 0,1,2,3,4,5,6,7,8,9,10,11,12,13,14,15);
}

// B-operand (32x16 bf16, KxN): lane<16 holds K = kb+0..15 ; lane>=16 holds K = kb+16..31.
DI v16bf load_frag_B(const __bf16* row, int kb, int lhalf) {
  v8bf lo = *(const v8bf*)(row + kb + 16 * lhalf);
  v8bf hi = *(const v8bf*)(row + kb + 16 * lhalf + 8);
  return __builtin_shufflevector(lo, hi, 0,1,2,3,4,5,6,7,8,9,10,11,12,13,14,15);
}

DI v8f wmma_bf16(v16bf a, v16bf b, v8f c) {
  return __builtin_amdgcn_wmma_f32_16x16x32_bf16(
      /*neg_a=*/false, a, /*neg_b=*/false, b,
      /*c_mod=*/(short)0, c, /*reuse_a=*/false, /*reuse_b=*/false);
}

// gfx1250 async global->LDS 16B copy (ASYNCcnt-tracked).
DI void async_cp16(void* lds_ptr, const void* gptr) {
  unsigned lds_off = (unsigned)(unsigned long long)(uintptr_t)lds_ptr;
  unsigned long long ga = (unsigned long long)(uintptr_t)gptr;
  asm volatile("global_load_async_to_lds_b128 %0, %1, off"
               :: "v"(lds_off), "v"(ga) : "memory");
}
DI void wait_async0() { asm volatile("s_wait_asynccnt 0x0" ::: "memory"); }

// ---------------- kernel 1: fp32 -> bf16 convert ----------------

__global__ void __launch_bounds__(256)
cvt_f32_bf16(const float* __restrict__ in, __bf16* __restrict__ out, int n4) {
  int i = blockIdx.x * 256 + threadIdx.x;
  if (i < n4) {
    float4 f = ((const float4*)in)[i];
    v4bf o;
    o[0] = f2bf(f.x); o[1] = f2bf(f.y); o[2] = f2bf(f.z); o[3] = f2bf(f.w);
    ((v4bf*)out)[i] = o;
  }
}

// ---------------- kernel 2: fused QKV projection ----------------
// Workgroup tile: 128 tokens x 128 features, K-step 64, double-buffered async LDS.
// Wave tile: 32 tokens x 64 features (8 accumulators, 16 WMMAs per k-step).
// Feature space 0..3071: [0,1024)->Q (scaled ch^-0.5), [1024,2048)->K, [2048,3072)->V.
// Q,K stored [b,h,n,ch] bf16; V stored transposed [b,h,ch,n] bf16.

__global__ void __launch_bounds__(256)
qkv_gemm(const __bf16* __restrict__ xb,
         const __bf16* __restrict__ wq, const __bf16* __restrict__ wk,
         const __bf16* __restrict__ wv,
         const float* __restrict__ bq, const float* __restrict__ bk,
         const float* __restrict__ bv,
         __bf16* __restrict__ Qh, __bf16* __restrict__ Kh, __bf16* __restrict__ Vt) {
  __shared__ __bf16 ldsA[2][128 * 64];   // [token][k]   16 KB per buffer
  __shared__ __bf16 ldsB[2][128 * 64];   // [feature][k] 16 KB per buffer

  const int tid   = threadIdx.x;
  const int lane  = tid & 31;
  const int wave  = tid >> 5;
  const int lmod  = lane & 15;
  const int lhalf = lane >> 4;

  const int mBase = blockIdx.x * 128;                    // 128 tokens per workgroup
  const int fg0   = blockIdx.y * 128;                    // 128 features in [0,3072)
  const int sel   = fg0 >> 10;                           // uniform per workgroup
  const int lf0   = fg0 & 1023;

  const __bf16* W    = (sel == 0) ? wq : (sel == 1) ? wk : wv;
  const float*  bias = (sel == 0) ? bq : (sel == 1) ? bk : bv;

  // per-thread async copy slots: 128 rows x 8 16B-chunks => 4 chunks each for A and B
  const int crow = tid >> 3, ccol = (tid & 7) * 8;

#pragma unroll
  for (int i = 0; i < 4; ++i) {
    async_cp16(&ldsA[0][(crow + 32 * i) * 64 + ccol],
               xb + (size_t)(mBase + crow + 32 * i) * E_ + ccol);
    async_cp16(&ldsB[0][(crow + 32 * i) * 64 + ccol],
               W + (size_t)(lf0 + crow + 32 * i) * E_ + ccol);
  }

  const int wm = wave & 3;                   // 32-token sub-block
  const int fW = (wave >> 2) * 64;           // 64-feature sub-block
  v8f acc[2][4] = {};
  int cur = 0;

  for (int k0 = 0; k0 < E_; k0 += 64) {
    wait_async0();          // own async copies for 'cur' have landed
    __syncthreads();        // everyone's copies landed; prev buffer reusable

    if (k0 + 64 < E_) {
      const int nxt = cur ^ 1;
#pragma unroll
      for (int i = 0; i < 4; ++i) {
        async_cp16(&ldsA[nxt][(crow + 32 * i) * 64 + ccol],
                   xb + (size_t)(mBase + crow + 32 * i) * E_ + k0 + 64 + ccol);
        async_cp16(&ldsB[nxt][(crow + 32 * i) * 64 + ccol],
                   W + (size_t)(lf0 + crow + 32 * i) * E_ + k0 + 64 + ccol);
      }
    }

    // hoist ALL fragments for this k-step, then fire 16 back-to-back WMMAs
    const __bf16* aRow0 = &ldsA[cur][(wm * 32 + lmod) * 64];
    const __bf16* aRow1 = aRow0 + 16 * 64;
    v16bf af[2][2], bf[2][4];
#pragma unroll
    for (int kk = 0; kk < 2; ++kk) {
      af[kk][0] = load_frag_A(aRow0, kk * 32, lhalf);
      af[kk][1] = load_frag_A(aRow1, kk * 32, lhalf);
#pragma unroll
      for (int g = 0; g < 4; ++g)
        bf[kk][g] = load_frag_B(&ldsB[cur][(fW + g * 16 + lmod) * 64], kk * 32, lhalf);
    }
#pragma unroll
    for (int kk = 0; kk < 2; ++kk)
#pragma unroll
      for (int g = 0; g < 4; ++g) {
        acc[0][g] = wmma_bf16(af[kk][0], bf[kk][g], acc[0][g]);
        acc[1][g] = wmma_bf16(af[kk][1], bf[kk][g], acc[1][g]);
      }
    cur ^= 1;
  }

#pragma unroll
  for (int r = 0; r < 2; ++r) {
    const int m0 = mBase + wm * 32 + r * 16;
    const int b  = m0 >> 11;
    const int n0 = m0 & (N_ - 1);
#pragma unroll
    for (int g = 0; g < 4; ++g) {
      const int fg = lf0 + fW + g * 16 + lmod;
      const float bb = bias[fg];
      const int h = fg >> 6, c = fg & 63;
      if (sel == 2) {
        v8bf pk;
#pragma unroll
        for (int i = 0; i < 8; ++i) pk[i] = f2bf(acc[r][g][i] + bb);
        *(v8bf*)(Vt + ((size_t)(b * H_ + h) * CH_ + c) * N_ + n0 + 8 * lhalf) = pk;
      } else {
        const float scale = (sel == 0) ? 0.125f : 1.0f;
        __bf16* dst = ((sel == 0) ? Qh : Kh) + (size_t)(b * H_ + h) * N_ * CH_;
#pragma unroll
        for (int i = 0; i < 8; ++i)
          dst[(size_t)(n0 + i + 8 * lhalf) * CH_ + c] = f2bf((acc[r][g][i] + bb) * scale);
      }
    }
  }
}

// ---------------- kernel 3: flash attention (transposed tiles) ----------------
// 8 waves share one (b,h); K/V blocks staged in LDS via async copies (8x traffic cut).
// S^T = K·Q^T, in-lane online softmax + one lane^16 shuffle, O^T = V^T·P^T.
// K and V fragments are hoisted together so V's LDS latency hides under softmax.

__global__ void __launch_bounds__(256)
attn_kernel(const __bf16* __restrict__ Qh, const __bf16* __restrict__ Kh,
            const __bf16* __restrict__ Vt, __bf16* __restrict__ Actx) {
  __shared__ __bf16 ldsK[2][32 * 64];   // [key][ch]  4 KB per buffer
  __shared__ __bf16 ldsV[2][64 * 32];   // [ch][key]  4 KB per buffer

  const int tid   = threadIdx.x;
  const int lane  = tid & 31;
  const int wave  = tid >> 5;
  const int lmod  = lane & 15;
  const int lhalf = lane >> 4;

  const int bh = blockIdx.x >> 4;            // b*H + h
  const int q0 = (blockIdx.x & 15) * 128 + wave * 16;

  const __bf16* Kb = Kh + (size_t)bh * N_ * CH_;
  const __bf16* Vb = Vt + (size_t)bh * CH_ * N_;

  // Q^T as B-operand, kept in registers for the whole kernel
  const __bf16* qrow = Qh + (size_t)bh * N_ * CH_ + (size_t)(q0 + lmod) * CH_;
  const v16bf qf0 = load_frag_B(qrow, 0,  lhalf);
  const v16bf qf1 = load_frag_B(qrow, 32, lhalf);

  // per-thread async copy slots
  const int krow = tid >> 3, kcol = (tid & 7) * 8;   // K 32x64
  const int vrow = tid >> 2, vcol = (tid & 3) * 8;   // V 64x32

  async_cp16(&ldsK[0][krow * 64 + kcol], Kb + (size_t)krow * CH_ + kcol);
  async_cp16(&ldsV[0][vrow * 32 + vcol], Vb + (size_t)vrow * N_ + vcol);

  v8f acc[4] = {};
  float mi = -3.0e38f, li = 0.0f;
  const v8f zero = {};
  int cur = 0;

  for (int j0 = 0; j0 < N_; j0 += 32) {
    wait_async0();
    __syncthreads();

    if (j0 + 32 < N_) {
      const int nxt = cur ^ 1;
      async_cp16(&ldsK[nxt][krow * 64 + kcol],
                 Kb + (size_t)(j0 + 32 + krow) * CH_ + kcol);
      async_cp16(&ldsV[nxt][vrow * 32 + vcol],
                 Vb + (size_t)vrow * N_ + j0 + 32 + vcol);
    }

    // hoist all K and V fragments for this block
    const __bf16* k0r = &ldsK[cur][lmod * 64];
    const __bf16* k1r = &ldsK[cur][(16 + lmod) * 64];
    v16bf kf00 = load_frag_A(k0r, 0,  lhalf);
    v16bf kf01 = load_frag_A(k0r, 32, lhalf);
    v16bf kf10 = load_frag_A(k1r, 0,  lhalf);
    v16bf kf11 = load_frag_A(k1r, 32, lhalf);
    v16bf vf[4];
#pragma unroll
    for (int t = 0; t < 4; ++t)
      vf[t] = load_frag_A(&ldsV[cur][(t * 16 + lmod) * 32], 0, lhalf);

    // S^T tiles: keys x queries
    v8f st0 = wmma_bf16(kf00, qf0, zero);
    st0     = wmma_bf16(kf01, qf1, st0);
    v8f st1 = wmma_bf16(kf10, qf0, zero);
    st1     = wmma_bf16(kf11, qf1, st1);

    // per-query softmax stats: 16 in-lane values + lane^16 partner
    float mb = st0[0];
#pragma unroll
    for (int i = 1; i < 8; ++i) mb = fmaxf(mb, st0[i]);
#pragma unroll
    for (int i = 0; i < 8; ++i) mb = fmaxf(mb, st1[i]);
    mb = fmaxf(mb, __shfl_xor(mb, 16, 32));
    const float mnew = fmaxf(mi, mb);

    float p0[8], p1[8], s = 0.0f;
#pragma unroll
    for (int i = 0; i < 8; ++i) {
      p0[i] = __expf(st0[i] - mnew); s += p0[i];
      p1[i] = __expf(st1[i] - mnew); s += p1[i];
    }
    s += __shfl_xor(s, 16, 32);
    const float alpha = __expf(mi - mnew);   // ==0 on first block
    li = li * alpha + s;
    mi = mnew;

    // P^T as B-operand: one packed shuffle per dword pair to fetch partner rows
    union { v16bf v; unsigned int u[8]; } pf;
#pragma unroll
    for (int j = 0; j < 4; ++j) {
      unsigned int u0 = (unsigned int)f2bfu(p0[2 * j]) |
                        ((unsigned int)f2bfu(p0[2 * j + 1]) << 16);
      unsigned int u1 = (unsigned int)f2bfu(p1[2 * j]) |
                        ((unsigned int)f2bfu(p1[2 * j + 1]) << 16);
      unsigned int s0 = (unsigned int)__shfl_xor((int)u0, 16, 32);
      unsigned int s1 = (unsigned int)__shfl_xor((int)u1, 16, 32);
      pf.u[j]     = lhalf ? s1 : u0;
      pf.u[4 + j] = lhalf ? u1 : s0;
    }

#pragma unroll
    for (int t = 0; t < 4; ++t) {
#pragma unroll
      for (int i = 0; i < 8; ++i) acc[t][i] *= alpha;
      acc[t] = wmma_bf16(vf[t], pf.v, acc[t]);
    }
    cur ^= 1;
  }

  const float inv = 1.0f / li;
  const int b = bh / H_, h = bh % H_;
  __bf16* orow = Actx + (size_t)(b * N_ + q0 + lmod) * E_ + h * CH_;
#pragma unroll
  for (int t = 0; t < 4; ++t) {
    v8bf pk;
#pragma unroll
    for (int i = 0; i < 8; ++i) pk[i] = f2bf(acc[t][i] * inv);
    *(v8bf*)(orow + t * 16 + 8 * lhalf) = pk;
  }
}

// ---------------- kernel 4: output projection (f32 out) ----------------

__global__ void __launch_bounds__(256)
out_gemm(const __bf16* __restrict__ Actx, const __bf16* __restrict__ wo,
         const float* __restrict__ bo, float* __restrict__ out) {
  __shared__ __bf16 ldsA[2][128 * 64];
  __shared__ __bf16 ldsB[2][128 * 64];

  const int tid   = threadIdx.x;
  const int lane  = tid & 31;
  const int wave  = tid >> 5;
  const int lmod  = lane & 15;
  const int lhalf = lane >> 4;

  const int mBase = blockIdx.x * 128;
  const int fg0   = blockIdx.y * 128;

  const int crow = tid >> 3, ccol = (tid & 7) * 8;

#pragma unroll
  for (int i = 0; i < 4; ++i) {
    async_cp16(&ldsA[0][(crow + 32 * i) * 64 + ccol],
               Actx + (size_t)(mBase + crow + 32 * i) * E_ + ccol);
    async_cp16(&ldsB[0][(crow + 32 * i) * 64 + ccol],
               wo + (size_t)(fg0 + crow + 32 * i) * E_ + ccol);
  }

  const int wm = wave & 3;
  const int fW = (wave >> 2) * 64;
  v8f acc[2][4] = {};
  int cur = 0;

  for (int k0 = 0; k0 < E_; k0 += 64) {
    wait_async0();
    __syncthreads();

    if (k0 + 64 < E_) {
      const int nxt = cur ^ 1;
#pragma unroll
      for (int i = 0; i < 4; ++i) {
        async_cp16(&ldsA[nxt][(crow + 32 * i) * 64 + ccol],
                   Actx + (size_t)(mBase + crow + 32 * i) * E_ + k0 + 64 + ccol);
        async_cp16(&ldsB[nxt][(crow + 32 * i) * 64 + ccol],
                   wo + (size_t)(fg0 + crow + 32 * i) * E_ + k0 + 64 + ccol);
      }
    }

    const __bf16* aRow0 = &ldsA[cur][(wm * 32 + lmod) * 64];
    const __bf16* aRow1 = aRow0 + 16 * 64;
    v16bf af[2][2], bf[2][4];
#pragma unroll
    for (int kk = 0; kk < 2; ++kk) {
      af[kk][0] = load_frag_A(aRow0, kk * 32, lhalf);
      af[kk][1] = load_frag_A(aRow1, kk * 32, lhalf);
#pragma unroll
      for (int g = 0; g < 4; ++g)
        bf[kk][g] = load_frag_B(&ldsB[cur][(fW + g * 16 + lmod) * 64], kk * 32, lhalf);
    }
#pragma unroll
    for (int kk = 0; kk < 2; ++kk)
#pragma unroll
      for (int g = 0; g < 4; ++g) {
        acc[0][g] = wmma_bf16(af[kk][0], bf[kk][g], acc[0][g]);
        acc[1][g] = wmma_bf16(af[kk][1], bf[kk][g], acc[1][g]);
      }
    cur ^= 1;
  }

#pragma unroll
  for (int r = 0; r < 2; ++r) {
    const int m0 = mBase + wm * 32 + r * 16;
#pragma unroll
    for (int g = 0; g < 4; ++g) {
      const int fg = fg0 + fW + g * 16 + lmod;
      const float bb = bo[fg];
#pragma unroll
      for (int i = 0; i < 8; ++i)
        out[(size_t)(m0 + i + 8 * lhalf) * E_ + fg] = acc[r][g][i] + bb;
    }
  }
}

// ---------------- host launch ----------------

extern "C" void kernel_launch(void* const* d_in, const int* in_sizes, int n_in,
                              void* d_out, int out_size, void* d_ws, size_t ws_size,
                              hipStream_t stream) {
  const float* x  = (const float*)d_in[0];
  const float* Wq = (const float*)d_in[1];
  const float* bq = (const float*)d_in[2];
  const float* Wk = (const float*)d_in[3];
  const float* bk = (const float*)d_in[4];
  const float* Wv = (const float*)d_in[5];
  const float* bv = (const float*)d_in[6];
  const float* Wo = (const float*)d_in[7];
  const float* bo = (const float*)d_in[8];
  float* out = (float*)d_out;

  // workspace carve-out (bf16), total ~92.3 MB
  char* p = (char*)d_ws;
  const size_t xN = (size_t)B_ * N_ * E_;   // 8388608
  const size_t wN = (size_t)E_ * E_;        // 1048576
  __bf16* xb  = (__bf16*)p;              p += xN * 2;
  __bf16* wqb = (__bf16*)p;              p += wN * 2;
  __bf16* wkb = (__bf16*)p;              p += wN * 2;
  __bf16* wvb = (__bf16*)p;              p += wN * 2;
  __bf16* wob = (__bf16*)p;              p += wN * 2;
  __bf16* Qh  = (__bf16*)p;              p += xN * 2;
  __bf16* Kh  = (__bf16*)p;              p += xN * 2;
  __bf16* Vt  = (__bf16*)p;              p += xN * 2;
  __bf16* Actx = (__bf16*)p;             p += xN * 2;

  cvt_f32_bf16<<<(int)(xN / 4 / 256), 256, 0, stream>>>(x, xb, (int)(xN / 4));
  cvt_f32_bf16<<<(int)(wN / 4 / 256), 256, 0, stream>>>(Wq, wqb, (int)(wN / 4));
  cvt_f32_bf16<<<(int)(wN / 4 / 256), 256, 0, stream>>>(Wk, wkb, (int)(wN / 4));
  cvt_f32_bf16<<<(int)(wN / 4 / 256), 256, 0, stream>>>(Wv, wvb, (int)(wN / 4));
  cvt_f32_bf16<<<(int)(wN / 4 / 256), 256, 0, stream>>>(Wo, wob, (int)(wN / 4));

  qkv_gemm<<<dim3(B_ * N_ / 128, 3 * E_ / 128), 256, 0, stream>>>(
      xb, wqb, wkb, wvb, bq, bk, bv, Qh, Kh, Vt);

  attn_kernel<<<B_ * H_ * (N_ / 128), 256, 0, stream>>>(Qh, Kh, Vt, Actx);

  out_gemm<<<dim3(B_ * N_ / 128, E_ / 128), 256, 0, stream>>>(Actx, wob, bo, out);
}
